// VJepaAlignerV3_7876970021314
// MI455X (gfx1250) — matrix operand order repaired and checked by reading, not compiled, not run on hardware
//
#include <hip/hip_runtime.h>
#include <hip/hip_bf16.h>
#include <math.h>

// ---------------- problem constants (from reference) ----------------
#define TS_     8
#define HS_     16
#define WSP_    16
#define COMMON_ 512
#define KPROTO  256
#define KMITERS 5
#define NB      16                 // batch B
#define NS_     (TS_*HS_*WSP_)     // 2048 student tokens
#define DS_     1024
#define DT_     1408
#define TT_     8
#define HT_     14
#define WT_     14
#define NT_     (TT_*HT_*WT_)      // 1568 teacher tokens
#define LN_EPS_   1e-5f
#define KM_EPS_   1e-6f
#define INV_TEMP_ 5.0f             // 1/0.2
#define GRAMW_    2.0f

typedef __attribute__((ext_vector_type(16))) __bf16 bf16x16;
typedef __attribute__((ext_vector_type(8)))  float  v8f;
typedef int v4i __attribute__((vector_size(16)));   // matches async-LDS builtin param

#if defined(__has_builtin)
#if __has_builtin(__builtin_amdgcn_global_load_async_to_lds_b128)
#define USE_ASYNC_LDS 1
#endif
#endif

#define WMMA_BF16(acc, a, b) \
    __builtin_amdgcn_wmma_f32_16x16x32_bf16(false, (a), false, (b), (short)0, (acc), false, false)

// ---------------- fragment helpers ----------------
// Packed B layout: per (kt,nt) tile, 32 lanes x 16 bf16 contiguous (32B/lane).
__device__ __forceinline__ bf16x16
load_b_packed(const __bf16* __restrict__ P, int tilesN, int kt, int nt, int lane)
{
    const __bf16* p = P + ((((size_t)kt * tilesN + nt) * 32 + lane) << 4);
    return *(const bf16x16*)p;
}

// Packed A layout: per (rt,kt) tile, 32 lanes x 16 bf16 contiguous.
__device__ __forceinline__ bf16x16
load_a_packed(const __bf16* __restrict__ P, size_t rtGlobal, int KT, int kt, int lane)
{
    const __bf16* p = P + (((rtGlobal * KT + kt) * 32 + lane) << 4);
    return *(const bf16x16*)p;
}

// B = C^T where C is row-major (KPROTO x COMMON): contiguous f32 + convert
__device__ __forceinline__ bf16x16
load_bT_rowmajor_f32(const float* __restrict__ C, int ld, int k0, int n0, int lane)
{
    const float* p = C + (size_t)(n0 + (lane & 15)) * ld + k0 + ((lane >> 4) << 3);
    bf16x16 b;
#pragma unroll
    for (int i = 0; i < 16; i += 4) {
        float4 f = *(const float4*)(p + i + ((i & 8) ? 8 : 0));
        b[i] = (__bf16)f.x; b[i+1] = (__bf16)f.y; b[i+2] = (__bf16)f.z; b[i+3] = (__bf16)f.w;
    }
    return b;
}

// ---------------- pack kernels ----------------
// Row-major X (batches x K x N) -> WMMA B-operand bf16 layout (per batch).
__global__ void k_pack_b(const float* __restrict__ X, __bf16* __restrict__ P,
                         size_t total, int K, int N)
{
    size_t tid = (size_t)blockIdx.x * blockDim.x + threadIdx.x;
    if (tid >= total) return;
    int n = (int)(tid % N);
    size_t rk = tid / N;
    int k  = (int)(rk % K);
    size_t bb = rk / K;
    int kt = k >> 5, half = (k >> 3) & 1, i = (k & 7) + ((k & 16) ? 8 : 0);
    int nt = n >> 4, nl = n & 15;
    int tilesN = N >> 4;
    size_t out = bb * (size_t)(K >> 5) * tilesN * 512
               + ((((size_t)kt * tilesN + nt) * 32) + (half * 16 + nl)) * 16 + i;
    P[out] = (__bf16)X[tid];
}

// Row-major X (Rtotal x K, rows contiguous per batch, R%16==0) -> A-operand layout.
__global__ void k_pack_a(const float* __restrict__ X, __bf16* __restrict__ P,
                         size_t total, int K)
{
    size_t tid = (size_t)blockIdx.x * blockDim.x + threadIdx.x;
    if (tid >= total) return;
    int k = (int)(tid % K);
    size_t r = tid / K;
    int kc = k & 31, kt = k >> 5;
    int half = (kc >> 3) & 1;
    int i = (kc & 7) + ((kc & 16) ? 8 : 0);
    int lanew = half * 16 + (int)(r & 15);
    size_t rt = r >> 4;
    size_t out = ((rt * (size_t)(K >> 5) + kt) * 32 + lanew) * 16 + i;
    P[out] = (__bf16)X[tid];
}

// w (NB x M x KPROTO row-major) -> A-operand layout of w^T (M=KPROTO rows, K=tokens)
__global__ void k_pack_aT(const float* __restrict__ W, __bf16* __restrict__ P, int M)
{
    size_t tid = (size_t)blockIdx.x * blockDim.x + threadIdx.x;
    size_t total = (size_t)NB * M * KPROTO;
    if (tid >= total) return;
    int p = (int)(tid % KPROTO);
    size_t tb = tid / KPROTO;
    int tok = (int)(tb % M);
    int bb  = (int)(tb / M);
    int kc = tok & 31, kt = tok >> 5;
    int half = (kc >> 3) & 1;
    int i = (kc & 7) + ((kc & 16) ? 8 : 0);
    int lanew = half * 16 + (p & 15);
    int rt = p >> 4;
    size_t out = ((((size_t)bb * (KPROTO / 16) + rt) * (M >> 5) + kt) * 32 + lanew) * 16 + i;
    P[out] = (__bf16)W[tid];
}

// ---------------- MLP GEMM: block-tiled, LDS-staged A (async) ----------------
// Block: 256 threads (8 waves). Block tile: 32(M) x 256(N).
// wave w: mtL = w&1 (m-tile), ng = w>>1 (group of 4 n-tiles).
#define NBLK 256
__global__ __launch_bounds__(256)
void k_gemm_bias(const float* __restrict__ X, const __bf16* __restrict__ P,
                 const float* __restrict__ bias, float* __restrict__ Y,
                 int M, int N, int K, int tilesN, int nHalves)
{
    __shared__ float sRaw[2][32 * 36];        // 32 rows x 32 k, stride 36 (bank pad)
    int t = threadIdx.x;
    int lane = t & 31, w = t >> 5;
    int mtL = w & 1;
    int ng  = w >> 1;
    int b  = blockIdx.y;
    int mb = blockIdx.x / nHalves;
    int nh = blockIdx.x % nHalves;
    const float* Xb = X + (size_t)b * M * K + (size_t)(mb * 32) * K;
    float*       Yb = Y + (size_t)b * M * N;
    int kTiles = K >> 5;
    v8f acc[4] = {{}, {}, {}, {}};

    int r_ = t >> 3, g_ = t & 7;              // staging role: row 0..31, float4 group 0..7
    auto stage = [&](int kt, int buf) {
        const float* src = Xb + (size_t)r_ * K + kt * 32 + g_ * 4;
        float* dst = &sRaw[buf][r_ * 36 + g_ * 4];
#ifdef USE_ASYNC_LDS
        __builtin_amdgcn_global_load_async_to_lds_b128((v4i*)src, (v4i*)dst, 0, 0);
#else
        *(float4*)dst = *(const float4*)src;
#endif
    };
    auto a_wait = [&]() {
#ifdef USE_ASYNC_LDS
        asm volatile("s_wait_asynccnt 0" ::: "memory");
#endif
    };

    stage(0, 0);
    for (int kt = 0; kt < kTiles; ++kt) {
        a_wait();
        __syncthreads();
        if (kt + 1 < kTiles) stage(kt + 1, (kt + 1) & 1);
        // gather this wave's A fragment from the raw f32 tile in LDS
        bf16x16 a;
        {
            int m = mtL * 16 + (lane & 15);
            int half = (lane >> 4) << 3;
            const float* base = &sRaw[kt & 1][m * 36];
#pragma unroll
            for (int v = 0; v < 8; ++v) {
                int kb = (v < 4) ? (half + 2 * v) : (16 + half + 2 * (v - 4));
                float2 f = *(const float2*)(base + kb);
                a[2 * v] = (__bf16)f.x; a[2 * v + 1] = (__bf16)f.y;
            }
        }
#pragma unroll
        for (int j = 0; j < 4; ++j) {
            int nt = nh * (NBLK / 16) + ng * 4 + j;
            bf16x16 bm = load_b_packed(P, tilesN, kt, nt, lane);
            acc[j] = WMMA_BF16(acc[j], a, bm);
        }
        __syncthreads();
    }
    int mrow = mb * 32 + mtL * 16 + ((lane >> 4) << 3);
#pragma unroll
    for (int j = 0; j < 4; ++j) {
        int n = (nh * (NBLK / 16) + ng * 4 + j) * 16 + (lane & 15);
        float bv = bias[n];
#pragma unroll
        for (int r = 0; r < 8; ++r) Yb[(size_t)(mrow + r) * N + n] = acc[j][r] + bv;
    }
}

// ---------------- k-means GEMM 1: Wb[b,n,k] = X . C^T  (4x4 reg-blocked) ----------
__global__ __launch_bounds__(32)
void k_gemm_xc4(const __bf16* __restrict__ Apk, const float* __restrict__ C,
                float* __restrict__ Wb, int M)
{
    int lane = threadIdx.x;
    int b = blockIdx.y;
    const int nGroups = KPROTO / 64;          // 4
    int mg = blockIdx.x / nGroups;            // 0 .. M/64-1
    int ng = blockIdx.x % nGroups;
    const float* Cb = C + (size_t)b * KPROTO * COMMON_;
    float*      Wbb = Wb + (size_t)b * M * KPROTO;
    const int KT = COMMON_ / 32;              // 16
    v8f acc[4][4];
#pragma unroll
    for (int i = 0; i < 4; ++i)
#pragma unroll
        for (int j = 0; j < 4; ++j) acc[i][j] = (v8f){};
    for (int kt = 0; kt < KT; ++kt) {
        bf16x16 a[4], bm[4];
#pragma unroll
        for (int i = 0; i < 4; ++i)
            a[i] = load_a_packed(Apk, (size_t)b * (M / 16) + mg * 4 + i, KT, kt, lane);
#pragma unroll
        for (int j = 0; j < 4; ++j)
            bm[j] = load_bT_rowmajor_f32(Cb, COMMON_, kt * 32, (ng * 4 + j) * 16, lane);
#pragma unroll
        for (int i = 0; i < 4; ++i)
#pragma unroll
            for (int j = 0; j < 4; ++j) acc[i][j] = WMMA_BF16(acc[i][j], a[i], bm[j]);
    }
#pragma unroll
    for (int i = 0; i < 4; ++i) {
        int mbr = mg * 64 + i * 16 + ((lane >> 4) << 3);
#pragma unroll
        for (int j = 0; j < 4; ++j) {
            int n = (ng * 4 + j) * 16 + (lane & 15);
#pragma unroll
            for (int r = 0; r < 8; ++r)
                Wbb[(size_t)(mbr + r) * KPROTO + n] = acc[i][j][r];
        }
    }
}

// ---------------- k-means GEMM 2: C = (w^T X) / (den+eps)  (4x4 reg-blocked) ------
__global__ __launch_bounds__(32)
void k_gemm_upd4(const __bf16* __restrict__ Atk, const __bf16* __restrict__ Bpk,
                 const float* __restrict__ Den, float* __restrict__ C, int M)
{
    int lane = threadIdx.x;
    int b = blockIdx.y;
    const int nGroups = COMMON_ / 64;         // 8
    int mg = blockIdx.x / nGroups;            // 0..3 (KPROTO/64)
    int ng = blockIdx.x % nGroups;
    const int KT = M >> 5;                    // token k-chunks (64)
    const int tilesN = COMMON_ / 16;          // 32
    v8f acc[4][4];
#pragma unroll
    for (int i = 0; i < 4; ++i)
#pragma unroll
        for (int j = 0; j < 4; ++j) acc[i][j] = (v8f){};
    const __bf16* Bb = Bpk + (size_t)b * (M >> 5) * tilesN * 512;
    for (int kt = 0; kt < KT; ++kt) {
        bf16x16 a[4], bm[4];
#pragma unroll
        for (int i = 0; i < 4; ++i)
            a[i] = load_a_packed(Atk, (size_t)b * (KPROTO / 16) + mg * 4 + i, KT, kt, lane);
#pragma unroll
        for (int j = 0; j < 4; ++j)
            bm[j] = load_b_packed(Bb, tilesN, kt, ng * 4 + j, lane);
#pragma unroll
        for (int i = 0; i < 4; ++i)
#pragma unroll
            for (int j = 0; j < 4; ++j) acc[i][j] = WMMA_BF16(acc[i][j], a[i], bm[j]);
    }
    float*       Cb = C + (size_t)b * KPROTO * COMMON_;
    const float* Dn = Den + b * KPROTO;
#pragma unroll
    for (int i = 0; i < 4; ++i) {
        int mbr = mg * 64 + i * 16 + ((lane >> 4) << 3);
#pragma unroll
        for (int j = 0; j < 4; ++j) {
            int n = (ng * 4 + j) * 16 + (lane & 15);
#pragma unroll
            for (int r = 0; r < 8; ++r)
                Cb[(size_t)(mbr + r) * COMMON_ + n] = acc[i][j][r] / (Dn[mbr + r] + KM_EPS_);
        }
    }
}

// ---------------- elementwise / reduction kernels ----------------
__global__ void k_ln_gelu(float* __restrict__ H, const float* __restrict__ g,
                          const float* __restrict__ beta)
{
    __shared__ float sm[256];
    float* row = H + (size_t)blockIdx.x * COMMON_;
    int t = threadIdx.x;
    float v0 = row[t], v1 = row[t + 256];
    sm[t] = v0 + v1; __syncthreads();
    for (int s = 128; s > 0; s >>= 1) { if (t < s) sm[t] += sm[t + s]; __syncthreads(); }
    float mu = sm[0] * (1.0f / COMMON_); __syncthreads();
    float d0 = v0 - mu, d1 = v1 - mu;
    sm[t] = d0 * d0 + d1 * d1; __syncthreads();
    for (int s = 128; s > 0; s >>= 1) { if (t < s) sm[t] += sm[t + s]; __syncthreads(); }
    float rstd = rsqrtf(sm[0] * (1.0f / COMMON_) + LN_EPS_);
    float h0 = d0 * rstd * g[t]       + beta[t];
    float h1 = d1 * rstd * g[t + 256] + beta[t + 256];
    row[t]       = 0.5f * h0 * (1.0f + erff(h0 * 0.70710678f));
    row[t + 256] = 0.5f * h1 * (1.0f + erff(h1 * 0.70710678f));
}

__global__ void k_resize(const float* __restrict__ T, float* __restrict__ O)
{
    size_t tid = (size_t)blockIdx.x * blockDim.x + threadIdx.x;
    size_t total = (size_t)NB * NS_ * COMMON_;
    if (tid >= total) return;
    int c  = (int)(tid % COMMON_);
    int ns = (int)((tid / COMMON_) % NS_);
    int b  = (int)(tid / ((size_t)COMMON_ * NS_));
    int xo = ns % WSP_, yo = (ns / WSP_) % HS_, zo = ns / (WSP_ * HS_);
    float fy = (yo + 0.5f) * (14.0f / 16.0f) - 0.5f;
    float fx = (xo + 0.5f) * (14.0f / 16.0f) - 0.5f;
    int y0 = (int)floorf(fy); float wy = fy - (float)y0;
    int x0 = (int)floorf(fx); float wx = fx - (float)x0;
    int y1 = y0 + 1, x1 = x0 + 1;
    y0 = y0 < 0 ? 0 : (y0 > 13 ? 13 : y0);  y1 = y1 < 0 ? 0 : (y1 > 13 ? 13 : y1);
    x0 = x0 < 0 ? 0 : (x0 > 13 ? 13 : x0);  x1 = x1 < 0 ? 0 : (x1 > 13 ? 13 : x1);
    const float* Tb = T + (size_t)b * NT_ * COMMON_;
    int zb = zo * HT_ * WT_;
    float a00 = Tb[(size_t)(zb + y0 * WT_ + x0) * COMMON_ + c];
    float a01 = Tb[(size_t)(zb + y0 * WT_ + x1) * COMMON_ + c];
    float a10 = Tb[(size_t)(zb + y1 * WT_ + x0) * COMMON_ + c];
    float a11 = Tb[(size_t)(zb + y1 * WT_ + x1) * COMMON_ + c];
    O[tid] = (1.0f - wy) * ((1.0f - wx) * a00 + wx * a01)
           +         wy  * ((1.0f - wx) * a10 + wx * a11);
}

__global__ void k_sqnorm(const float* __restrict__ X, float* __restrict__ O)
{
    __shared__ float sm[256];
    const float* row = X + (size_t)blockIdx.x * COMMON_;
    int t = threadIdx.x;
    float a = row[t], b = row[t + 256];
    sm[t] = a * a + b * b; __syncthreads();
    for (int s = 128; s > 0; s >>= 1) { if (t < s) sm[t] += sm[t + s]; __syncthreads(); }
    if (t == 0) O[blockIdx.x] = sm[0];
}

__global__ void k_gather_c(const float* __restrict__ X, const int* __restrict__ idx,
                           float* __restrict__ C, int M)
{
    int tid = blockIdx.x * blockDim.x + threadIdx.x;
    int total = NB * KPROTO * COMMON_;
    if (tid >= total) return;
    int d = tid % COMMON_;
    int k = (tid / COMMON_) % KPROTO;
    int b = tid / (COMMON_ * KPROTO);
    int row = idx[b * KPROTO + k];
    C[tid] = X[((size_t)b * M + row) * COMMON_ + d];
}

__global__ void k_softmax(float* __restrict__ Wb, const float* __restrict__ X2,
                          const float* __restrict__ C2, int M)
{
    __shared__ float sm[256];
    int b = blockIdx.x / M, n = blockIdx.x % M;
    int k = threadIdx.x;
    float* row = Wb + ((size_t)b * M + n) * KPROTO;
    float d2 = X2[b * M + n] + C2[b * KPROTO + k] - 2.0f * row[k];
    float logit = -d2 * INV_TEMP_;
    sm[k] = logit; __syncthreads();
    for (int s = 128; s > 0; s >>= 1) { if (k < s) sm[k] = fmaxf(sm[k], sm[k + s]); __syncthreads(); }
    float mx = sm[0]; __syncthreads();
    float e = __expf(logit - mx);
    sm[k] = e; __syncthreads();
    for (int s = 128; s > 0; s >>= 1) { if (k < s) sm[k] += sm[k + s]; __syncthreads(); }
    row[k] = e / sm[0];
}

__global__ void k_denom(const float* __restrict__ Wb, float* __restrict__ Den, int M)
{
    int b = blockIdx.x, k = threadIdx.x;
    const float* base = Wb + (size_t)b * M * KPROTO;
    float s = 0.0f;
    for (int n = 0; n < M; ++n) s += base[(size_t)n * KPROTO + k];
    Den[b * KPROTO + k] = s;
}

__global__ void k_zero(float* p) { p[0] = 0.0f; }

__global__ void k_loss(const float* __restrict__ A, const float* __restrict__ Bv,
                       float* __restrict__ out, int n, float scale)
{
    __shared__ float sm[256];
    float s = 0.0f;
    for (int i = blockIdx.x * blockDim.x + threadIdx.x; i < n; i += gridDim.x * blockDim.x) {
        float d = A[i] - Bv[i];
        s += d * d;
    }
    sm[threadIdx.x] = s; __syncthreads();
    for (int r = 128; r > 0; r >>= 1) {
        if (threadIdx.x < r) sm[threadIdx.x] += sm[threadIdx.x + r];
        __syncthreads();
    }
    if (threadIdx.x == 0) atomicAdd(out, sm[0] * scale);
}

// ---------------- host orchestration ----------------
extern "C" void kernel_launch(void* const* d_in, const int* in_sizes, int n_in,
                              void* d_out, int out_size, void* d_ws, size_t ws_size,
                              hipStream_t stream)
{
    const float* student = (const float*)d_in[0];
    const float* teacher = (const float*)d_in[1];
    const int*   idx_s   = (const int*)d_in[2];
    const int*   idx_t   = (const int*)d_in[3];
    const float* Ws1 = (const float*)d_in[7];
    const float* bs1 = (const float*)d_in[8];
    const float* gs  = (const float*)d_in[9];
    const float* bes = (const float*)d_in[10];
    const float* Ws2 = (const float*)d_in[11];
    const float* bs2 = (const float*)d_in[12];
    const float* Wt1 = (const float*)d_in[13];
    const float* bt1 = (const float*)d_in[14];
    const float* gt  = (const float*)d_in[15];
    const float* bet = (const float*)d_in[16];
    const float* Wt2 = (const float*)d_in[17];
    const float* bt2 = (const float*)d_in[18];
    float* out = (float*)d_out;
    (void)ws_size; (void)n_in; (void)in_sizes; (void)out_size;

    char* base = (char*)d_ws;
    size_t off = 0;
    auto carve = [&](size_t bytes) -> void* {
        void* p = base + off;
        off = (off + bytes + 255) & ~(size_t)255;
        return p;
    };
    __bf16* wpS1 = (__bf16*)carve((size_t)DS_ * COMMON_ * 2);
    __bf16* wpS2 = (__bf16*)carve((size_t)COMMON_ * COMMON_ * 2);
    __bf16* wpT1 = (__bf16*)carve((size_t)DT_ * COMMON_ * 2);
    __bf16* wpT2 = (__bf16*)carve((size_t)COMMON_ * COMMON_ * 2);
    float* bufA = (float*)carve((size_t)NB * NS_ * COMMON_ * 4);   // h_s, later t_tok
    float* bufS = (float*)carve((size_t)NB * NS_ * COMMON_ * 4);   // s
    float* bufB = (float*)carve((size_t)NB * NT_ * COMMON_ * 4);   // h_t, later w
    float* bufT = (float*)carve((size_t)NB * NT_ * COMMON_ * 4);   // t
    float* cS   = (float*)carve((size_t)NB * KPROTO * COMMON_ * 4);
    float* cT   = (float*)carve((size_t)NB * KPROTO * COMMON_ * 4);
    __bf16* sApk = (__bf16*)carve((size_t)NB * NS_ * COMMON_ * 2);
    __bf16* sBpk = (__bf16*)carve((size_t)NB * NS_ * COMMON_ * 2);
    __bf16* tApk = (__bf16*)carve((size_t)NB * NS_ * COMMON_ * 2);
    __bf16* tBpk = (__bf16*)carve((size_t)NB * NS_ * COMMON_ * 2);
    __bf16* wTpk = (__bf16*)carve((size_t)NB * NS_ * KPROTO * 2);
    float* x2S  = (float*)carve((size_t)NB * NS_ * 4);
    float* x2T  = (float*)carve((size_t)NB * NS_ * 4);
    float* c2b  = (float*)carve((size_t)NB * KPROTO * 4);
    float* den  = (float*)carve((size_t)NB * KPROTO * 4);

    const int tilesN = COMMON_ / 16;    // 32
    const int nHalves = COMMON_ / NBLK; // 2

    // pack weights into WMMA-B layout (treated as 1-batch K x N matrices)
    k_pack_b<<<(DS_ * COMMON_ + 255) / 256, 256, 0, stream>>>(Ws1, wpS1, (size_t)DS_ * COMMON_, DS_, COMMON_);
    k_pack_b<<<(COMMON_ * COMMON_ + 255) / 256, 256, 0, stream>>>(Ws2, wpS2, (size_t)COMMON_ * COMMON_, COMMON_, COMMON_);
    k_pack_b<<<(DT_ * COMMON_ + 255) / 256, 256, 0, stream>>>(Wt1, wpT1, (size_t)DT_ * COMMON_, DT_, COMMON_);
    k_pack_b<<<(COMMON_ * COMMON_ + 255) / 256, 256, 0, stream>>>(Wt2, wpT2, (size_t)COMMON_ * COMMON_, COMMON_, COMMON_);

    // student MLP
    k_gemm_bias<<<dim3((NS_ / 32) * nHalves, NB), 256, 0, stream>>>(
        student, wpS1, bs1, bufA, NS_, COMMON_, DS_, tilesN, nHalves);
    k_ln_gelu<<<NB * NS_, 256, 0, stream>>>(bufA, gs, bes);
    k_gemm_bias<<<dim3((NS_ / 32) * nHalves, NB), 256, 0, stream>>>(
        bufA, wpS2, bs2, bufS, NS_, COMMON_, COMMON_, tilesN, nHalves);

    // teacher MLP
    k_gemm_bias<<<dim3((NT_ / 32) * nHalves, NB), 256, 0, stream>>>(
        teacher, wpT1, bt1, bufB, NT_, COMMON_, DT_, tilesN, nHalves);
    k_ln_gelu<<<NB * NT_, 256, 0, stream>>>(bufB, gt, bet);
    k_gemm_bias<<<dim3((NT_ / 32) * nHalves, NB), 256, 0, stream>>>(
        bufB, wpT2, bt2, bufT, NT_, COMMON_, COMMON_, tilesN, nHalves);

    // resize teacher tokens -> t_tok (reuses bufA)
    {
        size_t tot = (size_t)NB * NS_ * COMMON_;
        k_resize<<<(unsigned)((tot + 255) / 256), 256, 0, stream>>>(bufT, bufA);
    }

    // pack x operands for k-means GEMMs (once per side)
    {
        size_t tot = (size_t)NB * NS_ * COMMON_;
        unsigned g = (unsigned)((tot + 255) / 256);
        k_pack_a<<<g, 256, 0, stream>>>(bufS, sApk, tot, COMMON_);
        k_pack_b<<<g, 256, 0, stream>>>(bufS, sBpk, tot, NS_, COMMON_);
        k_pack_a<<<g, 256, 0, stream>>>(bufA, tApk, tot, COMMON_);
        k_pack_b<<<g, 256, 0, stream>>>(bufA, tBpk, tot, NS_, COMMON_);
    }

    // soft k-means (bufB reused as assignment matrix w)
    auto kmeans = [&](const float* x, const __bf16* xApk, const __bf16* xBpk,
                      const int* idx, float* x2, float* c) {
        k_sqnorm<<<NB * NS_, 256, 0, stream>>>(x, x2);
        {
            int tot = NB * KPROTO * COMMON_;
            k_gather_c<<<(tot + 255) / 256, 256, 0, stream>>>(x, idx, c, NS_);
        }
        for (int it = 0; it < KMITERS; ++it) {
            k_sqnorm<<<NB * KPROTO, 256, 0, stream>>>(c, c2b);
            k_gemm_xc4<<<dim3((NS_ / 64) * (KPROTO / 64), NB), 32, 0, stream>>>(xApk, c, bufB, NS_);
            k_softmax<<<NB * NS_, 256, 0, stream>>>(bufB, x2, c2b, NS_);
            k_denom<<<NB, KPROTO, 0, stream>>>(bufB, den, NS_);
            {
                size_t tot = (size_t)NB * NS_ * KPROTO;
                k_pack_aT<<<(unsigned)((tot + 255) / 256), 256, 0, stream>>>(bufB, wTpk, NS_);
            }
            k_gemm_upd4<<<dim3((KPROTO / 64) * (COMMON_ / 64), NB), 32, 0, stream>>>(
                wTpk, xBpk, den, c, NS_);
        }
    };
    kmeans(bufS, sApk, sBpk, idx_s, x2S, cS);
    kmeans(bufA, tApk, tBpk, idx_t, x2T, cT);

    // loss
    k_zero<<<1, 1, 0, stream>>>(out);
    {
        int n = NB * KPROTO * COMMON_;
        float scale = GRAMW_ / (float)n;
        k_loss<<<512, 256, 0, stream>>>(cS, cT, out, n, scale);
    }
}
